// DepthConv_82703890252333
// MI455X (gfx1250) — compile-verified, matrix operand order checked by reading, not run on hardware
//
#include <hip/hip_runtime.h>

typedef _Float16 half_t;
typedef __attribute__((ext_vector_type(16))) _Float16 v16h;
typedef __attribute__((ext_vector_type(8)))  float    v8f;

namespace {
constexpr int   kBn    = 8;
constexpr int   kC     = 16;
constexpr int   kH     = 256;
constexpr int   kW     = 256;
constexpr int   kO     = 16;
constexpr int   kOH    = 254;
constexpr int   kOW    = 254;
// exp(-8.3*d) == exp2(-8.3*log2(e)*d): fold into one constant, feed v_exp_f32
constexpr float kNegAlphaLog2e = -8.3f * 1.4426950408889634f;
constexpr int   kRows  = 2;                   // output rows per block (254 = 2*127)
constexpr int   kYG    = kOH / kRows;         // 127 row-groups
constexpr int   kBlocks = kBn * kYG;          // 1016 blocks
constexpr int   kCell  = 24;                  // halves per cell (48B): 16 used + pad
                                              // 48B stride -> 12-bank lane stride ->
                                              // 16 lanes * 4 banks tile all 64 banks
}

__global__ __launch_bounds__(256)
void depthconv_wmma_kernel(const float* __restrict__ data,
                           const float* __restrict__ depth,
                           const float* __restrict__ weight,
                           float* __restrict__ out)
{
  // 4 data rows (f16, channel-contiguous) + packed A-layout weights + 4 depth rows
  __shared__ __align__(16) _Float16 smemD[4 * kW * kCell];   // 49152 B
  __shared__ __align__(16) _Float16 smemW[5 * 32 * kCell];   //  7680 B
  __shared__ __align__(16) float    smemZ[4 * kW];           //  4096 B

  const int tid  = threadIdx.x;
  const int wave = tid >> 5;
  const int lane = tid & 31;
  const int n    = lane & 15;   // pixel column within tile; also A-row m
  const int khf  = lane >> 4;   // K-half selector (wave32 WMMA layout)

  const int yg = blockIdx.x % kYG;
  const int b  = blockIdx.x / kYG;
  const int y0 = yg * kRows;    // first output row; needs input rows y0..y0+3

  const float* dat = data  + b * (kC * kH * kW);
  const float* dep = depth + b * (kH * kW);

  // ---- stage 4 data rows: f32 -> f16 exactly once per element ----
  // thread t owns x = t; global loads lane-coalesced over x; 16 contiguous
  // half stores merge into ds_store_b128 pairs (48B cell, 16B aligned).
#pragma unroll
  for (int r = 0; r < 4; ++r) {
    const float* src = dat + (y0 + r) * kW + tid;
    _Float16* dst = smemD + (r * kW + tid) * kCell;
#pragma unroll
    for (int c = 0; c < kC; ++c) dst[c] = (half_t)src[c * (kH * kW)];
  }

  // ---- stage 4 depth rows ----
#pragma unroll
  for (int r = 0; r < 4; ++r) smemZ[r * kW + tid] = dep[(y0 + r) * kW + tid];

  // ---- stage weights in packed A-fragment layout, converted + zero-padded ----
  // element e = ((t*32 + L) << 4) | s : chunk t, lane L, slot s (= channel c).
  // K-order kg = kidx*16 + c, kidx = 2t + (L>>4); kidx==9 => zero pad.
#pragma unroll
  for (int j = 0; j < 10; ++j) {
    const int e    = tid + 256 * j;        // 0..2559
    const int s    = e & 15;
    const int L    = (e >> 4) & 31;
    const int t    = e >> 9;
    const int wn   = L & 15;
    const int wkh  = L >> 4;
    const int kidx = 2 * t + wkh;
    const float wv = (kidx < 9) ? weight[(wn * kC + s) * 9 + kidx] : 0.0f;
    smemW[(t * 32 + L) * kCell + s] = (half_t)wv;
  }

  __syncthreads();

  // ---- A fragments: direct v16h loads (2x ds_load_b128 each) ----
  v16h afr[5];
#pragma unroll
  for (int t = 0; t < 5; ++t) {
    afr[t] = *(const v16h*)(smemW + (t * 32 + lane) * kCell);
  }

  // ---- compute: 2 output rows x 2 tiles of 16 pixels per wave ----
#pragma unroll
  for (int yy = 0; yy < kRows; ++yy) {
    const int y = y0 + yy;
#pragma unroll
    for (int i = 0; i < 2; ++i) {
      const int  x0    = (wave * 2 + i) * 16;
      const int  x     = x0 + n;
      const bool colOK = (x < kOW);
      const int  xs    = colOK ? x : 0;   // clamped column for predicated-off lanes

      // -- affinities from LDS: aff[kidx] = exp2(c * |z_center - z_k|) --
      const float zc = smemZ[(yy + 1) * kW + (xs + 1)];
      float aff9[9];
#pragma unroll
      for (int ki = 0; ki < 3; ++ki) {
#pragma unroll
        for (int kj = 0; kj < 3; ++kj) {
          float z = smemZ[(yy + ki) * kW + (xs + kj)];
          aff9[ki * 3 + kj] =
              __builtin_amdgcn_exp2f(kNegAlphaLog2e * fabsf(zc - z));
        }
      }

      // -- K loop: 5 chained V_WMMA_F32_16X16X32_F16 --
      v8f acc = {};
#pragma unroll
      for (int t = 0; t < 5; ++t) {
        const int  kidx = 2 * t + khf;
        const bool kOK  = (kidx < 9) && colOK;

        const int kLo = 2 * t;                                 // khf==0 tap
        const int kHi = (2 * t + 1 < 9) ? (2 * t + 1) : 8;     // khf==1 tap (clamped)
        const int ki  = khf ? (kHi / 3) : (kLo / 3);
        const int kj  = khf ? (kHi % 3) : (kLo % 3);

        const float affLo = aff9[kLo];
        const float affHi = (2 * t + 1 < 9) ? aff9[(2 * t + 1) % 9] : 0.0f;
        // aff = 0 zeroes pad chunk and tail columns (reads stay clamped-safe)
        const float aff   = khf ? affHi : affLo;
        const half_t ah   = (half_t)(kOK ? aff : 0.0f);

        v16h aff16;
#pragma unroll
        for (int s = 0; s < 16; ++s) aff16[s] = ah;

        // B fragment: 16 contiguous f16 channels at (row yy+ki, col xs+kj),
        // loaded directly as v16h (2x ds_load_b128 into adjacent VGPRs)
        const v16h raw =
            *(const v16h*)(smemD + (((yy + ki) * kW) + (xs + kj)) * kCell);

        const v16h bfr = raw * aff16;   // 8x v_pk_mul_f16

        acc = __builtin_amdgcn_wmma_f32_16x16x32_f16(
            /*neg_a=*/false, afr[t], /*neg_b=*/false, bfr,
            /*c_mod=*/(short)0, acc, /*reuse_a=*/false, /*reuse_b=*/false);
      }

      // -- store D (streaming, non-temporal): VGPR r -> row m = r + 8*khf --
      if (colOK) {
#pragma unroll
        for (int r = 0; r < 8; ++r) {
          const int mrow = r + 8 * khf;
          __builtin_nontemporal_store(acc[r],
              &out[((b * kO + mrow) * kOH + y) * kOW + x]);
        }
      }
    }
  }
}

extern "C" void kernel_launch(void* const* d_in, const int* /*in_sizes*/, int /*n_in*/,
                              void* d_out, int /*out_size*/, void* /*d_ws*/, size_t /*ws_size*/,
                              hipStream_t stream) {
  const float* data   = (const float*)d_in[0];
  const float* depth  = (const float*)d_in[1];
  const float* weight = (const float*)d_in[2];
  float* out = (float*)d_out;

  hipLaunchKernelGGL(depthconv_wmma_kernel,
                     dim3(kBlocks), dim3(256), 0, stream,
                     data, depth, weight, out);
}
